// FAdapter_19920058319069
// MI455X (gfx1250) — matrix operand order repaired and checked by reading, not compiled, not run on hardware
//
#include <hip/hip_runtime.h>

// B=16, S=2048, D=768, K=64. LN over (S,D) per batch, then
// out = x + relu(xn @ W_down^T) @ W_up^T.  Memory-bound (~300MB traffic,
// ~13us at 23.3 TB/s); 6.4 GFLOP is trivial for the bf16 WMMA pipe.
//
// ws layout (floats): [0,4096) partial sums (2048 blocks x {sum,sumsq})
//                     [4096,4128) per-batch {mu, rstd}
//                     [4128, ...) packed bf16 B-fragments: W_down (24576 dwords)
//                                 then W_up (24576 dwords).  ~209 KB total.

typedef float v8f __attribute__((ext_vector_type(8)));
typedef __bf16 v16bf __attribute__((ext_vector_type(16)));

union FragAB {
    unsigned u[8];
    uint4    q[2];
    v16bf    v;
};

__device__ __forceinline__ unsigned short f32_to_bf16(float f) {
    unsigned u = __float_as_uint(f);
    unsigned r = u + 0x7FFFu + ((u >> 16) & 1u);   // round-to-nearest-even
    return (unsigned short)(r >> 16);
}
__device__ __forceinline__ unsigned pack_bf16x2(float lo, float hi) {
    return (unsigned)f32_to_bf16(lo) | ((unsigned)f32_to_bf16(hi) << 16);
}

// ---------------- Pass 1: per-chunk partial sum / sumsq --------------------
__global__ __launch_bounds__(256) void ln_partial(const float* __restrict__ x,
                                                  float* __restrict__ partials) {
    // 128 chunks per batch, 12288 elements each
    int b = blockIdx.x >> 7;
    int c = blockIdx.x & 127;
    const float4* xv = (const float4*)(x + (size_t)b * (2048u * 768u) + (size_t)c * 12288u);
    float s = 0.f, q = 0.f;
    for (int i = threadIdx.x; i < 3072; i += 256) {
        float4 v = xv[i];
        s += v.x + v.y + v.z + v.w;
        q += v.x * v.x + v.y * v.y + v.z * v.z + v.w * v.w;
    }
    __shared__ float ss[256], sq[256];
    ss[threadIdx.x] = s; sq[threadIdx.x] = q;
    __syncthreads();
    for (int o = 128; o > 0; o >>= 1) {
        if (threadIdx.x < o) {
            ss[threadIdx.x] += ss[threadIdx.x + o];
            sq[threadIdx.x] += sq[threadIdx.x + o];
        }
        __syncthreads();
    }
    if (threadIdx.x == 0) {
        partials[blockIdx.x * 2 + 0] = ss[0];
        partials[blockIdx.x * 2 + 1] = sq[0];
    }
}

// ---------------- Pass 2: finalize per-batch mu / rstd ---------------------
__global__ __launch_bounds__(128) void ln_finalize(const float* __restrict__ partials,
                                                   float* __restrict__ stats) {
    int b = blockIdx.x, t = threadIdx.x;
    __shared__ float ss[128], sq[128];
    ss[t] = partials[(b * 128 + t) * 2 + 0];
    sq[t] = partials[(b * 128 + t) * 2 + 1];
    __syncthreads();
    for (int o = 64; o > 0; o >>= 1) {
        if (t < o) { ss[t] += ss[t + o]; sq[t] += sq[t + o]; }
        __syncthreads();
    }
    if (t == 0) {
        const float invN = 1.0f / (2048.0f * 768.0f);
        float mu  = ss[0] * invN;
        float var = sq[0] * invN - mu * mu;
        stats[b * 2 + 0] = mu;
        stats[b * 2 + 1] = rsqrtf(var + 1e-5f);
    }
}

// ------------- Pass 3: pack weights into bf16 WMMA B-fragments -------------
// B-fragment packing used here (contiguous k, same order as the A loads): lane L -> N = L%16, half = L/16;
// dword v (0..7) holds K = ktile*32 + 16*half + 2*v (low), +1 (high).
__global__ __launch_bounds__(256) void pack_weights(const float* __restrict__ Wd,  // [64,768]
                                                    const float* __restrict__ Wu,  // [768,64]
                                                    unsigned* __restrict__ wdp,
                                                    unsigned* __restrict__ wup) {
    int idx = blockIdx.x * 256 + threadIdx.x;           // 49152 total
    if (idx < 24576) {
        // GEMM1: B[d][n] = W_down[n*768 + d]; frag index = ((kt*4+nt)*32+lane)*8+v
        int v = idx & 7, lane = (idx >> 3) & 31;
        int nt = (idx >> 8) & 3, kt = idx >> 10;
        int half = lane >> 4, n = nt * 16 + (lane & 15);
        int kk = kt * 32 + 16 * half + 2 * v;
        wdp[idx] = pack_bf16x2(Wd[n * 768 + kk], Wd[n * 768 + kk + 1]);
    } else {
        // GEMM2: B[k][n] = W_up[n*64 + k]; frag index = ((kt*48+nt)*32+lane)*8+v
        int j = idx - 24576;
        int v = j & 7, lane = (j >> 3) & 31;
        int nt = (j >> 8) % 48, kt = (j >> 8) / 48;
        int half = lane >> 4, n = nt * 16 + (lane & 15);
        int kk = kt * 32 + 16 * half + 2 * v;
        wup[j] = pack_bf16x2(Wu[n * 64 + kk], Wu[n * 64 + kk + 1]);
    }
}

// ---------------- Pass 4: fused normalize + GEMM1 + ReLU + GEMM2 -----------
#define XPITCH 1552   // 768 bf16 = 1536 B + 16 B pad (bank-conflict-free b128)
#define HPITCH 144    // 64 bf16 = 128 B + 16 B pad

__global__ __launch_bounds__(128) void fused_mlp(
    const float* __restrict__ x, const float* __restrict__ Wln,
    const float* __restrict__ bln, const unsigned* __restrict__ wdp,
    const unsigned* __restrict__ wup, const float* __restrict__ stats,
    float* __restrict__ out) {
    __shared__ __align__(16) unsigned char s_xn[64 * XPITCH];  // 97 KB bf16 xn tile
    __shared__ __align__(16) unsigned char s_h[64 * HPITCH];   //  9 KB bf16 h tile

    const int tid = threadIdx.x;
    const int wave = tid >> 5, lane = tid & 31;
    const int half = lane >> 4, lcol = lane & 15;
    const int batch = blockIdx.x >> 5;                 // 32 blocks per batch
    const int row0 = (blockIdx.x & 31) * 64;           // 64 rows per block
    const size_t gbase = ((size_t)batch * 2048 + row0) * 768;
    const float mu = stats[batch * 2 + 0];
    const float inv = stats[batch * 2 + 1];

    // ---- normalize x tile -> bf16 in LDS ----
    const float4* xv = (const float4*)(x + gbase);
    const float4* wv = (const float4*)Wln;
    const float4* bv = (const float4*)bln;
    for (int i = tid; i < 64 * 192; i += 128) {
        int r = i / 192, dq = i - r * 192;
        float4 xx = xv[r * 192 + dq];
        float4 w = wv[dq], bb = bv[dq];
        float y0 = (xx.x - mu) * inv * w.x + bb.x;
        float y1 = (xx.y - mu) * inv * w.y + bb.y;
        float y2 = (xx.z - mu) * inv * w.z + bb.z;
        float y3 = (xx.w - mu) * inv * w.w + bb.w;
        uint2 p;
        p.x = pack_bf16x2(y0, y1);
        p.y = pack_bf16x2(y2, y3);
        *(uint2*)(s_xn + r * XPITCH + dq * 8) = p;
    }
    __syncthreads();

    const int m0 = wave * 16;  // each wave owns 16 rows

    // ---- GEMM1: h[16,64] = xn[16,768] @ W_down^T, K-tiles of 32 ----
    // A-fragment as loaded here (contiguous k, same order as B): lane -> M = lane%16; two contiguous 16B
    // chunks per lane at byte offsets half*32 and half*32+16 within k-tile.
    v8f acc0 = {0,0,0,0,0,0,0,0}, acc1 = acc0, acc2 = acc0, acc3 = acc0;
    const unsigned char* arow = s_xn + (m0 + lcol) * XPITCH + half * 32;
    for (int kt = 0; kt < 24; ++kt) {
        FragAB a;
        a.q[0] = *(const uint4*)(arow + kt * 64);
        a.q[1] = *(const uint4*)(arow + kt * 64 + 16);
        const uint4* bp = (const uint4*)(wdp + ((size_t)(kt * 4) * 32 + lane) * 8);
        FragAB b0, b1, b2, b3;
        b0.q[0] = bp[0];  b0.q[1] = bp[1];
        b1.q[0] = bp[64]; b1.q[1] = bp[65];   // +32 lanes * 32B = 64 uint4
        b2.q[0] = bp[128]; b2.q[1] = bp[129];
        b3.q[0] = bp[192]; b3.q[1] = bp[193];
        acc0 = __builtin_amdgcn_wmma_f32_16x16x32_bf16(false, a.v, false, b0.v, (short)0, acc0, false, false);
        acc1 = __builtin_amdgcn_wmma_f32_16x16x32_bf16(false, a.v, false, b1.v, (short)0, acc1, false, false);
        acc2 = __builtin_amdgcn_wmma_f32_16x16x32_bf16(false, a.v, false, b2.v, (short)0, acc2, false, false);
        acc3 = __builtin_amdgcn_wmma_f32_16x16x32_bf16(false, a.v, false, b3.v, (short)0, acc3, false, false);
    }

    // ---- ReLU + bf16 h tile to LDS (C layout: row = r + 8*half, col = lcol) ----
    {
        v8f accs[4] = {acc0, acc1, acc2, acc3};
        #pragma unroll
        for (int nt = 0; nt < 4; ++nt) {
            #pragma unroll
            for (int r = 0; r < 8; ++r) {
                float v = accs[nt][r];
                v = v > 0.f ? v : 0.f;
                *(unsigned short*)(s_h + (m0 + r + 8 * half) * HPITCH +
                                   (nt * 16 + lcol) * 2) = f32_to_bf16(v);
            }
        }
    }
    __syncthreads();

    // ---- GEMM2: out[16,768] = x + h[16,64] @ W_up^T ----
    FragAB a2[2];
    const unsigned char* hrow = s_h + (m0 + lcol) * HPITCH + half * 32;
    #pragma unroll
    for (int kt = 0; kt < 2; ++kt) {
        a2[kt].q[0] = *(const uint4*)(hrow + kt * 64);
        a2[kt].q[1] = *(const uint4*)(hrow + kt * 64 + 16);
    }
    for (int nt = 0; nt < 48; ++nt) {
        v8f c = {0,0,0,0,0,0,0,0};
        #pragma unroll
        for (int kt = 0; kt < 2; ++kt) {
            FragAB bfr;
            const uint4* bp = (const uint4*)(wup + ((size_t)(kt * 48 + nt) * 32 + lane) * 8);
            bfr.q[0] = bp[0];
            bfr.q[1] = bp[1];
            c = __builtin_amdgcn_wmma_f32_16x16x32_bf16(false, a2[kt].v, false, bfr.v, (short)0, c, false, false);
        }
        const int col = nt * 16 + lcol;
        const size_t base2 = gbase + (size_t)(m0 + 8 * half) * 768 + col;
        #pragma unroll
        for (int r = 0; r < 8; ++r) {
            size_t idx = base2 + (size_t)r * 768;
            out[idx] = x[idx] + c[r];
        }
    }
}

extern "C" void kernel_launch(void* const* d_in, const int* in_sizes, int n_in,
                              void* d_out, int out_size, void* d_ws, size_t ws_size,
                              hipStream_t stream) {
    const float* x     = (const float*)d_in[0];
    const float* W_up  = (const float*)d_in[1];   // [768, 64]
    const float* W_down = (const float*)d_in[3];  // [64, 768]
    const float* W_ln  = (const float*)d_in[5];   // [768]
    const float* b_ln  = (const float*)d_in[6];   // [768]
    float* out = (float*)d_out;

    float* wsf = (float*)d_ws;
    float* partials = wsf;                        // 4096 floats
    float* stats = wsf + 4096;                    // 32 floats
    unsigned* wdp = (unsigned*)(wsf + 4128);      // 24576 dwords (16B aligned)
    unsigned* wup = wdp + 24576;                  // 24576 dwords

    ln_partial<<<2048, 256, 0, stream>>>(x, partials);
    ln_finalize<<<16, 128, 0, stream>>>(partials, stats);
    pack_weights<<<192, 256, 0, stream>>>(W_down, W_up, wdp, wup);
    fused_mlp<<<512, 128, 0, stream>>>(x, W_ln, b_ln, wdp, wup, stats, out);
}
